// CrossScaleFusion_11957188952173
// MI455X (gfx1250) — matrix operand order repaired and checked by reading, not compiled, not run on hardware
//
#include <hip/hip_runtime.h>

#define Hd   128
#define Bdim 32
#define Ndim 4096
#define Cdim 512
#define LSTR 132          // f32 LDS row stride (dwords): conflict-free, 16B-aligned rows
#define ASTR 264          // bf16 LDS row stride (elements): same dword stride as LSTR
#define TILEF (16 * LSTR) // floats per 16-row f32 tile buffer

typedef __attribute__((ext_vector_type(16))) __bf16 v16bf;
typedef __attribute__((ext_vector_type(8)))  __bf16 v8bf;
typedef __attribute__((ext_vector_type(8)))  float  v8f;

__device__ __forceinline__ __bf16 f2bf(float f) {
  unsigned int u = __float_as_uint(f);
  u += 0x7FFFu + ((u >> 16) & 1u);          // round-to-nearest-even
  unsigned short s = (unsigned short)(u >> 16);
  __bf16 r;
  __builtin_memcpy(&r, &s, sizeof(r));
  return r;
}

__device__ __forceinline__ v8bf pack8(float4 a, float4 b) {
  v8bf o;
  o[0] = f2bf(a.x); o[1] = f2bf(a.y); o[2] = f2bf(a.z); o[3] = f2bf(a.w);
  o[4] = f2bf(b.x); o[5] = f2bf(b.y); o[6] = f2bf(b.z); o[7] = f2bf(b.w);
  return o;
}

__device__ __forceinline__ void zero_acc(v8f acc[8]) {
  v8f z = {0.f, 0.f, 0.f, 0.f, 0.f, 0.f, 0.f, 0.f};
#pragma unroll
  for (int i = 0; i < 8; ++i) acc[i] = z;
}

// D(16x128) += A(16x(32*ktiles) bf16 in LDS, row-major stride ASTR) x W (pre-swizzled bf16)
// A fragment (16-bit A 16x32, contiguous k order matching packw, not the ISA table order): lane L = row m=L%16; halves at k-offsets kb and kb+8, kb=(L/16)*16
// -> two 16B ds loads + register concat.
__device__ __forceinline__ void mm_acc(const __bf16* __restrict__ Ab,
                                       const unsigned short* __restrict__ Wp,
                                       int ktiles, int lane, v8f acc[8]) {
  const v16bf* wf = reinterpret_cast<const v16bf*>(Wp);
  const int m  = lane & 15;
  const int kb = (lane >> 4) * 16;
  const __bf16* arow = Ab + m * ASTR + kb;
#pragma unroll 1
  for (int kt = 0; kt < ktiles; ++kt) {
    const __bf16* p = arow + kt * 32;
    v8bf lo = *(const v8bf*)(p);
    v8bf hi = *(const v8bf*)(p + 8);
    v16bf a = __builtin_shufflevector(lo, hi, 0, 1, 2, 3, 4, 5, 6, 7,
                                      8, 9, 10, 11, 12, 13, 14, 15);
#pragma unroll
    for (int nt = 0; nt < 8; ++nt) {
      v16bf b = wf[(kt * 8 + nt) * 32 + lane];
      acc[nt] = __builtin_amdgcn_wmma_f32_16x16x32_bf16(
          false, a, false, b, (short)0, acc[nt], false, false);
    }
  }
}

// D layout (f32 16x16): lane L holds n=L%16, rows m = v + (L/16)*8
__device__ __forceinline__ void store_acc_bias(float* __restrict__ dst,
                                               const float* __restrict__ bias,
                                               int lane, const v8f acc[8]) {
  const int half = lane >> 4, nlo = lane & 15;
#pragma unroll
  for (int nt = 0; nt < 8; ++nt)
#pragma unroll
    for (int v = 0; v < 8; ++v) {
      int mm = v + half * 8, nn = nt * 16 + nlo;
      dst[mm * LSTR + nn] = acc[nt][v] + bias[nn];
    }
}

// LayerNorm+ReLU over 16 rows; lane pairs (r, r+16) split a row's 128 cols.
// Writes normalized f32 back and a bf16 copy into Ab+aoff (matmul A staging).
__device__ __forceinline__ void ln_relu_rows(float* __restrict__ buf,
                                             __bf16* __restrict__ Ab, int aoff,
                                             const float* __restrict__ g,
                                             const float* __restrict__ be, int lane) {
  const int r = lane & 15;
  const int c0 = (lane >> 4) * 64;
  float* row = buf + r * LSTR;
  float s = 0.f;
#pragma unroll
  for (int c = 0; c < 64; c += 4) {
    float4 f = *(const float4*)(row + c0 + c);
    s += f.x + f.y + f.z + f.w;
  }
  s += __shfl_xor(s, 16);
  float mu = s * (1.f / Hd);
  float var = 0.f;
#pragma unroll
  for (int c = 0; c < 64; c += 4) {
    float4 f = *(const float4*)(row + c0 + c);
    float dx = f.x - mu, dy = f.y - mu, dz = f.z - mu, dw = f.w - mu;
    var += dx * dx + dy * dy + dz * dz + dw * dw;
  }
  var += __shfl_xor(var, 16);
  float rs = rsqrtf(var * (1.f / Hd) + 1e-5f);
#pragma unroll
  for (int c = 0; c < 64; c += 8) {
    int cc = c0 + c;
    float4 f0 = *(const float4*)(row + cc);
    float4 f1 = *(const float4*)(row + cc + 4);
    f0.x = fmaxf((f0.x - mu) * rs * g[cc + 0] + be[cc + 0], 0.f);
    f0.y = fmaxf((f0.y - mu) * rs * g[cc + 1] + be[cc + 1], 0.f);
    f0.z = fmaxf((f0.z - mu) * rs * g[cc + 2] + be[cc + 2], 0.f);
    f0.w = fmaxf((f0.w - mu) * rs * g[cc + 3] + be[cc + 3], 0.f);
    f1.x = fmaxf((f1.x - mu) * rs * g[cc + 4] + be[cc + 4], 0.f);
    f1.y = fmaxf((f1.y - mu) * rs * g[cc + 5] + be[cc + 5], 0.f);
    f1.z = fmaxf((f1.z - mu) * rs * g[cc + 6] + be[cc + 6], 0.f);
    f1.w = fmaxf((f1.w - mu) * rs * g[cc + 7] + be[cc + 7], 0.f);
    *(float4*)(row + cc) = f0;
    *(float4*)(row + cc + 4) = f1;
    *(v8bf*)(Ab + r * ASTR + aoff + cc) = pack8(f0, f1);
  }
}

// gate = sigmoid(acc+b); Pd = gate*Pf + (1-gate)*Px ; also bf16 copy into Ab+aoff
__device__ __forceinline__ void gate_fuse(const v8f acc[8], const float* __restrict__ bg,
                                          const float* __restrict__ Pf,
                                          const float* __restrict__ Px,
                                          float* __restrict__ Pd,
                                          __bf16* __restrict__ Ab, int aoff, int lane) {
  const int half = lane >> 4, nlo = lane & 15;
#pragma unroll
  for (int nt = 0; nt < 8; ++nt)
#pragma unroll
    for (int v = 0; v < 8; ++v) {
      int mm = v + half * 8, nn = nt * 16 + nlo;
      float z  = acc[nt][v] + bg[nn];
      float gt = 1.f / (1.f + __expf(-z));
      float fv = Pf[mm * LSTR + nn], xv = Px[mm * LSTR + nn];
      float o  = gt * fv + (1.f - gt) * xv;
      Pd[mm * LSTR + nn] = o;
      Ab[mm * ASTR + aoff + nn] = f2bf(o);
    }
}

// f32 LDS tile -> bf16 A staging
__device__ __forceinline__ void lds_f32_to_bf16(const float* __restrict__ Pf,
                                                __bf16* __restrict__ Ab, int aoff, int lane) {
  for (int ch = lane; ch < 256; ch += 32) {
    int r = ch >> 4, c = (ch & 15) * 8;
    float4 f0 = *(const float4*)(Pf + r * LSTR + c);
    float4 f1 = *(const float4*)(Pf + r * LSTR + c + 4);
    *(v8bf*)(Ab + r * ASTR + aoff + c) = pack8(f0, f1);
  }
}

// ---------------- prep kernels ----------------

__global__ void packw(const float* __restrict__ W, unsigned short* __restrict__ Wp, int Ktot) {
  int p = blockIdx.x * blockDim.x + threadIdx.x;
  if (p >= Ktot * Hd) return;
  int e = p & 15, lane = (p >> 4) & 31, nt = (p >> 9) & 7, kt = p >> 12;
  int k = kt * 32 + ((lane >> 4) << 4) + e;
  int n = (nt << 4) + (lane & 15);
  unsigned int u = __float_as_uint(W[k * Hd + n]);
  u += 0x7FFFu + ((u >> 16) & 1u);
  Wp[p] = (unsigned short)(u >> 16);
}

__global__ void zero_ws(float* __restrict__ p, int n) {
  int i = blockIdx.x * blockDim.x + threadIdx.x;
  if (i < n) p[i] = 0.f;
}

__global__ void scatter_pool(const float* __restrict__ fine, const int* __restrict__ idx,
                             float* __restrict__ pooled, float* __restrict__ counts) {
  size_t i = (size_t)blockIdx.x * blockDim.x + threadIdx.x;  // over B*N*H
  int c    = (int)(i & (Hd - 1));
  size_t r = i >> 7;            // row in [0, B*N)
  int b    = (int)(r >> 12);    // N = 4096
  int iv   = idx[r];
  if (iv >= 0 && iv < Cdim) {
    atomicAdd(&pooled[(((size_t)b * Cdim + iv) << 7) + c], fine[i]);
    if (c == 0) atomicAdd(&counts[(size_t)b * Cdim + iv], 1.0f);
  }
}

__global__ void gmean_kernel(const float* __restrict__ glob, float* __restrict__ gmean) {
  int b = blockIdx.x, h = threadIdx.x;
  const float* p = glob + (size_t)b * Ndim * Hd + h;
  float s = 0.f;
  for (int n = 0; n < Ndim; ++n) s += p[(size_t)n * Hd];
  gmean[b * Hd + h] = s * (1.f / Ndim);
}

// ---------------- fused fine path (1 wave / block, one 16-row tile) ----------------

__global__ void __launch_bounds__(32) fuse_fine(
    const float* __restrict__ fine, const float* __restrict__ coarse,
    const float* __restrict__ glob, const int* __restrict__ idx,
    const float* __restrict__ b_un, const float* __restrict__ g_un, const float* __restrict__ be_un,
    const float* __restrict__ b_gate,
    const float* __restrict__ b_gl, const float* __restrict__ g_gl, const float* __restrict__ be_gl,
    const unsigned short* __restrict__ WpU, const unsigned short* __restrict__ WpG,
    const unsigned short* __restrict__ WpGl, float* __restrict__ out) {
  __shared__ __attribute__((aligned(16))) float  bufF[TILEF];   // fine rows (f32)
  __shared__ __attribute__((aligned(16))) float  bufX[TILEF];   // matmul out / x2 / wg
  __shared__ __attribute__((aligned(16))) float  bufU[TILEF];   // fused rows (fu0)
  __shared__ __attribute__((aligned(16))) __bf16 Abuf[16 * ASTR]; // 16x256 bf16 A staging
  const int lane = threadIdx.x;
  const int row0 = blockIdx.x * 16;
  const int b    = row0 >> 12;     // N = 4096; tiles never span batches

  // stage: gathered coarse -> Abuf[0:128] (bf16); fine -> bufF (f32)
  for (int ch = lane; ch < 256; ch += 32) {
    int r = ch >> 4, c = (ch & 15) * 8;
    const float* fs = fine + (size_t)(row0 + r) * Hd + c;
    float4 f0 = *(const float4*)(fs);
    float4 f1 = *(const float4*)(fs + 4);
    *(float4*)(bufF + r * LSTR + c)     = f0;
    *(float4*)(bufF + r * LSTR + c + 4) = f1;
    int iv = idx[row0 + r];
    float4 g0 = {0.f, 0.f, 0.f, 0.f}, g1 = {0.f, 0.f, 0.f, 0.f};
    if (iv >= 0 && iv < Cdim) {
      const float* cs = coarse + ((size_t)b * Cdim + iv) * Hd + c;
      g0 = *(const float4*)(cs);
      g1 = *(const float4*)(cs + 4);
    }
    *(v8bf*)(Abuf + r * ASTR + c) = pack8(g0, g1);
  }
  __syncthreads();

  v8f acc[8];
  zero_acc(acc);
  mm_acc(Abuf, WpU, 4, lane, acc);                 // unpool: gathered @ W_unpool
  store_acc_bias(bufX, b_un, lane, acc);
  __syncthreads();
  lds_f32_to_bf16(bufF, Abuf, 0, lane);            // Abuf[0:128] := bf16(fine)
  ln_relu_rows(bufX, Abuf, 128, g_un, be_un, lane); // x2 -> bufX(f32) + Abuf[128:256]
  __syncthreads();

  zero_acc(acc);
  mm_acc(Abuf, WpG, 8, lane, acc);                 // [fine | x2] @ W_gate
  gate_fuse(acc, b_gate, bufF, bufX, bufU, Abuf, 0, lane); // fu0 -> bufU + Abuf[0:128]
  for (int ch = lane; ch < 256; ch += 32) {        // global rows -> Abuf[128:256]
    int r = ch >> 4, c = (ch & 15) * 8;
    const float* gs = glob + (size_t)(row0 + r) * Hd + c;
    *(v8bf*)(Abuf + r * ASTR + 128 + c) = pack8(*(const float4*)(gs), *(const float4*)(gs + 4));
  }
  __syncthreads();

  zero_acc(acc);
  mm_acc(Abuf, WpGl, 8, lane, acc);                // [fu0 | global] @ W_glob
  store_acc_bias(bufX, b_gl, lane, acc);
  __syncthreads();
  ln_relu_rows(bufX, Abuf, 128, g_gl, be_gl, lane);
  __syncthreads();

  for (int ch = lane; ch < 256; ch += 32) {
    int r = ch >> 4, c = (ch & 15) * 8;
    float4 u0 = *(const float4*)(bufU + r * LSTR + c);
    float4 u1 = *(const float4*)(bufU + r * LSTR + c + 4);
    float4 w0 = *(const float4*)(bufX + r * LSTR + c);
    float4 w1 = *(const float4*)(bufX + r * LSTR + c + 4);
    float4 o0 = {u0.x + 0.1f * w0.x, u0.y + 0.1f * w0.y, u0.z + 0.1f * w0.z, u0.w + 0.1f * w0.w};
    float4 o1 = {u1.x + 0.1f * w1.x, u1.y + 0.1f * w1.y, u1.z + 0.1f * w1.z, u1.w + 0.1f * w1.w};
    float* os = out + (size_t)(row0 + r) * Hd + c;
    *(float4*)(os)     = o0;
    *(float4*)(os + 4) = o1;
  }
}

// ---------------- fused coarse path ----------------

__global__ void __launch_bounds__(32) fuse_coarse(
    const float* __restrict__ coarse, const float* __restrict__ pooled,
    const float* __restrict__ counts, const float* __restrict__ gmean,
    const float* __restrict__ b_p, const float* __restrict__ g_p, const float* __restrict__ be_p,
    const float* __restrict__ b_gate,
    const float* __restrict__ b_gl, const float* __restrict__ g_gl, const float* __restrict__ be_gl,
    const unsigned short* __restrict__ WpP, const unsigned short* __restrict__ WpG,
    const unsigned short* __restrict__ WpGl, float* __restrict__ out) {
  __shared__ __attribute__((aligned(16))) float  bufF[TILEF];
  __shared__ __attribute__((aligned(16))) float  bufX[TILEF];
  __shared__ __attribute__((aligned(16))) float  bufU[TILEF];
  __shared__ __attribute__((aligned(16))) __bf16 Abuf[16 * ASTR];
  const int lane = threadIdx.x;
  const int row0 = blockIdx.x * 16;                // row in [0, B*C)
  const int b    = row0 >> 9;                      // C = 512

  // stage: pooled mean -> Abuf[0:128] (bf16); coarse rows -> bufF (f32)
  for (int ch = lane; ch < 256; ch += 32) {
    int r = ch >> 4, c = (ch & 15) * 8;
    const float* cs = coarse + (size_t)(row0 + r) * Hd + c;
    *(float4*)(bufF + r * LSTR + c)     = *(const float4*)(cs);
    *(float4*)(bufF + r * LSTR + c + 4) = *(const float4*)(cs + 4);
    float inv = 1.f / fmaxf(counts[row0 + r], 1.f);
    const float* ps = pooled + (size_t)(row0 + r) * Hd + c;
    float4 p0 = *(const float4*)(ps);
    float4 p1 = *(const float4*)(ps + 4);
    p0.x *= inv; p0.y *= inv; p0.z *= inv; p0.w *= inv;
    p1.x *= inv; p1.y *= inv; p1.z *= inv; p1.w *= inv;
    *(v8bf*)(Abuf + r * ASTR + c) = pack8(p0, p1);
  }
  __syncthreads();

  v8f acc[8];
  zero_acc(acc);
  mm_acc(Abuf, WpP, 4, lane, acc);                 // pooled_mean @ W_pool
  store_acc_bias(bufX, b_p, lane, acc);
  __syncthreads();
  lds_f32_to_bf16(bufF, Abuf, 0, lane);            // Abuf[0:128] := bf16(coarse)
  ln_relu_rows(bufX, Abuf, 128, g_p, be_p, lane);  // x1 -> bufX + Abuf[128:256]
  __syncthreads();

  zero_acc(acc);
  mm_acc(Abuf, WpG, 8, lane, acc);                 // [coarse | x1] @ W_gate
  gate_fuse(acc, b_gate, bufF, bufX, bufU, Abuf, 0, lane); // upd -> bufU + Abuf[0:128]
  for (int ch = lane; ch < 256; ch += 32) {        // broadcast gmean -> Abuf[128:256]
    int r = ch >> 4, c = (ch & 15) * 8;
    const float* gs = gmean + b * Hd + c;
    *(v8bf*)(Abuf + r * ASTR + 128 + c) = pack8(*(const float4*)(gs), *(const float4*)(gs + 4));
  }
  __syncthreads();

  zero_acc(acc);
  mm_acc(Abuf, WpGl, 8, lane, acc);                // [upd | gmean] @ W_glob
  store_acc_bias(bufX, b_gl, lane, acc);
  __syncthreads();
  ln_relu_rows(bufX, Abuf, 128, g_gl, be_gl, lane);
  __syncthreads();

  for (int ch = lane; ch < 256; ch += 32) {
    int r = ch >> 4, c = (ch & 15) * 8;
    float4 u0 = *(const float4*)(bufU + r * LSTR + c);
    float4 u1 = *(const float4*)(bufU + r * LSTR + c + 4);
    float4 w0 = *(const float4*)(bufX + r * LSTR + c);
    float4 w1 = *(const float4*)(bufX + r * LSTR + c + 4);
    float4 o0 = {u0.x + 0.1f * w0.x, u0.y + 0.1f * w0.y, u0.z + 0.1f * w0.z, u0.w + 0.1f * w0.w};
    float4 o1 = {u1.x + 0.1f * w1.x, u1.y + 0.1f * w1.y, u1.z + 0.1f * w1.z, u1.w + 0.1f * w1.w};
    float* os = out + (size_t)(row0 + r) * Hd + c;
    *(float4*)(os)     = o0;
    *(float4*)(os + 4) = o1;
  }
}

// ---------------- launch ----------------

extern "C" void kernel_launch(void* const* d_in, const int* in_sizes, int n_in,
                              void* d_out, int out_size, void* d_ws, size_t ws_size,
                              hipStream_t stream) {
  const float* fine   = (const float*)d_in[0];
  const float* coarse = (const float*)d_in[1];
  const float* glob   = (const float*)d_in[2];
  const int*   idx    = (const int*)d_in[3];
  const float* W_pool = (const float*)d_in[4];
  const float* b_pool = (const float*)d_in[5];
  const float* g_pool = (const float*)d_in[6];
  const float* be_pool= (const float*)d_in[7];
  const float* W_un   = (const float*)d_in[8];
  const float* b_un   = (const float*)d_in[9];
  const float* g_un   = (const float*)d_in[10];
  const float* be_un  = (const float*)d_in[11];
  const float* W_gate = (const float*)d_in[12];
  const float* b_gate = (const float*)d_in[13];
  const float* W_gl   = (const float*)d_in[14];
  const float* b_gl   = (const float*)d_in[15];
  const float* g_gl   = (const float*)d_in[16];
  const float* be_gl  = (const float*)d_in[17];

  // workspace layout
  float* pooled = (float*)d_ws;                                   // B*C*H
  float* counts = pooled + (size_t)Bdim * Cdim * Hd;              // B*C
  float* gmean  = counts + (size_t)Bdim * Cdim;                   // B*H
  unsigned short* wpPool = (unsigned short*)(gmean + (size_t)Bdim * Hd);
  unsigned short* wpUn   = wpPool + (size_t)Hd * Hd;
  unsigned short* wpGate = wpUn + (size_t)Hd * Hd;
  unsigned short* wpGlob = wpGate + (size_t)2 * Hd * Hd;

  // weight packing (bf16, WMMA B-fragment order)
  packw<<<(Hd * Hd + 255) / 256, 256, 0, stream>>>(W_pool, wpPool, Hd);
  packw<<<(Hd * Hd + 255) / 256, 256, 0, stream>>>(W_un, wpUn, Hd);
  packw<<<(2 * Hd * Hd + 255) / 256, 256, 0, stream>>>(W_gate, wpGate, 2 * Hd);
  packw<<<(2 * Hd * Hd + 255) / 256, 256, 0, stream>>>(W_gl, wpGlob, 2 * Hd);

  // pooling + global mean
  int zn = Bdim * Cdim * Hd + Bdim * Cdim;
  zero_ws<<<(zn + 255) / 256, 256, 0, stream>>>(pooled, zn);
  scatter_pool<<<(int)(((size_t)Bdim * Ndim * Hd) / 256), 256, 0, stream>>>(fine, idx, pooled, counts);
  gmean_kernel<<<Bdim, Hd, 0, stream>>>(glob, gmean);

  float* out_fine   = (float*)d_out;
  float* out_coarse = out_fine + (size_t)Bdim * Ndim * Hd;

  fuse_fine<<<Bdim * Ndim / 16, 32, 0, stream>>>(
      fine, coarse, glob, idx,
      b_un, g_un, be_un, b_gate, b_gl, g_gl, be_gl,
      wpUn, wpGate, wpGlob, out_fine);

  fuse_coarse<<<Bdim * Cdim / 16, 32, 0, stream>>>(
      coarse, pooled, counts, gmean,
      b_pool, g_pool, be_pool, b_gate, b_gl, g_gl, be_gl,
      wpPool, wpGate, wpGlob, out_coarse);
}
